// Transformer_Fusion_20478404067522
// MI455X (gfx1250) — compile-verified, matrix operand order checked by reading, not run on hardware
//
#include <hip/hip_runtime.h>

#define B_ 2
#define C_ 32
#define H_ 80
#define W_ 80
#define L_ (H_ * W_)                 // 6400
#define KD 288                       // C*9
#define KSTEPS 9                     // 288 / 32
#define NCHUNK 8
#define MTILES (L_ / 16)             // 400
#define LT_PER_CHUNK (MTILES / NCHUNK) // 50

typedef __attribute__((ext_vector_type(16))) __bf16 v16bf;
typedef __attribute__((ext_vector_type(8)))  __bf16 v8bf;
typedef __attribute__((ext_vector_type(8)))  float  v8f;

__device__ __forceinline__ unsigned short f2bf(float f) {
  unsigned int u = __float_as_uint(f);
  u += 0x7FFFu + ((u >> 16) & 1u);   // round-to-nearest-even
  return (unsigned short)(u >> 16);
}

// Monotone float <-> uint32 order-preserving map (for branch-free max+argmax).
__device__ __forceinline__ unsigned int fmap(float f) {
  unsigned int u = __float_as_uint(f);
  return (u & 0x80000000u) ? ~u : (u | 0x80000000u);
}
__device__ __forceinline__ float funmap(unsigned int m) {
  unsigned int u = (m & 0x80000000u) ? (m & 0x7FFFFFFFu) : ~m;
  return __uint_as_float(u);
}

// ---------------------------------------------------------------------------
// Kernel 1: build column-L2-normalized unfold matrices as bf16, layout [b, l, k]
// ---------------------------------------------------------------------------
__global__ __launch_bounds__(256)
void tf_build_norm(const float* __restrict__ HSI, const float* __restrict__ PAN,
                   unsigned short* __restrict__ HSIn, unsigned short* __restrict__ PANn) {
  int idx = blockIdx.x * blockDim.x + threadIdx.x;
  if (idx >= B_ * L_) return;
  int b = idx / L_, l = idx - b * L_;
  int y = l / W_, x = l - (l / W_) * W_;

  float sH = 0.f, sP = 0.f;
  for (int c = 0; c < C_; ++c) {
    const float* hb = HSI + (size_t)(b * C_ + c) * L_;
    const float* pb = PAN + (size_t)(b * C_ + c) * L_;
#pragma unroll
    for (int dy = -1; dy <= 1; ++dy)
#pragma unroll
      for (int dx = -1; dx <= 1; ++dx) {
        int yy = y + dy, xx = x + dx;
        bool ok = (yy >= 0) & (yy < H_) & (xx >= 0) & (xx < W_);
        float hv = ok ? hb[yy * W_ + xx] : 0.f;
        float pv = ok ? pb[yy * W_ + xx] : 0.f;
        sH += hv * hv; sP += pv * pv;
      }
  }
  float iH = 1.f / fmaxf(sqrtf(sH), 1e-12f);
  float iP = 1.f / fmaxf(sqrtf(sP), 1e-12f);

  unsigned short* ho = HSIn + (size_t)idx * KD;
  unsigned short* po = PANn + (size_t)idx * KD;
  for (int c = 0; c < C_; ++c) {
    const float* hb = HSI + (size_t)(b * C_ + c) * L_;
    const float* pb = PAN + (size_t)(b * C_ + c) * L_;
#pragma unroll
    for (int dy = -1; dy <= 1; ++dy)
#pragma unroll
      for (int dx = -1; dx <= 1; ++dx) {
        int yy = y + dy, xx = x + dx;
        bool ok = (yy >= 0) & (yy < H_) & (xx >= 0) & (xx < W_);
        float hv = ok ? hb[yy * W_ + xx] : 0.f;
        float pv = ok ? pb[yy * W_ + xx] : 0.f;
        int k = c * 9 + (dy + 1) * 3 + (dx + 1);
        ho[k] = f2bf(hv * iH);
        po[k] = f2bf(pv * iP);
      }
  }
}

// ---------------------------------------------------------------------------
// Kernel 2: Rt[m,l] = sum_k HSIn[m,k]*PANn[l,k] via bf16 WMMA with fused
// branch-free running max/argmax. Double-buffered B tiles (ping-pong regs).
// ---------------------------------------------------------------------------
__device__ __forceinline__ v16bf load_tile16(const unsigned short* p) {
  v8bf lo = *(const v8bf*)(p);
  v8bf hi = *(const v8bf*)(p + 16);
  v16bf r;
#pragma unroll
  for (int i = 0; i < 8; ++i) { r[i] = lo[i]; r[i + 8] = hi[i]; }
  return r;
}

__device__ __forceinline__ void loadB9(const unsigned short* base, int koff, v16bf B[KSTEPS]) {
#pragma unroll
  for (int s = 0; s < KSTEPS; ++s)
    B[s] = load_tile16(base + s * 32 + koff);
}

__device__ __forceinline__ void tile_compute(const v16bf A[KSTEPS], const v16bf B[KSTEPS],
                                             int lbase, int rowcol,
                                             unsigned long long rkey[8]) {
  v8f c = {};
#pragma unroll
  for (int s = 0; s < KSTEPS; ++s)
    c = __builtin_amdgcn_wmma_f32_16x16x32_bf16(false, A[s], false, B[s],
                                                (short)0, c, false, false);
  // Branch-free per-lane running (max, first-argmax): pack value (monotone map)
  // in high 32 bits, ~idx in low 32 bits -> u64 max == (max val, lowest idx).
  const unsigned int lo = ~(unsigned int)(lbase + rowcol);
#pragma unroll
  for (int r = 0; r < 8; ++r) {
    unsigned long long key = ((unsigned long long)fmap(c[r]) << 32) | (unsigned long long)lo;
    rkey[r] = (key > rkey[r]) ? key : rkey[r];
  }
}

__global__ __launch_bounds__(256)
void tf_gemm_argmax(const unsigned short* __restrict__ HSIn,
                    const unsigned short* __restrict__ PANn,
                    float* __restrict__ pMax, int* __restrict__ pArg) {
  const int lane    = threadIdx.x & 31;
  const int waveId  = blockIdx.x * (blockDim.x >> 5) + (threadIdx.x >> 5);
  const int b       = waveId / (MTILES * NCHUNK);
  const int rem     = waveId - b * (MTILES * NCHUNK);
  const int mtile   = rem / NCHUNK;
  const int chunk   = rem - mtile * NCHUNK;
  const int mbase   = mtile * 16;

  const int halfsel = lane >> 4;      // 0 or 1
  const int rowcol  = lane & 15;      // M for A-load, N for B-load
  const int koff    = halfsel * 8;    // ISA bf16 operand layout: k0 = 0 / 8

  // Preload A: HSIn rows mbase..mbase+15, full K=288 (reused for all 50 l-tiles)
  const unsigned short* Abase = HSIn + ((size_t)b * L_ + mbase) * KD;
  v16bf Areg[KSTEPS];
#pragma unroll
  for (int s = 0; s < KSTEPS; ++s)
    Areg[s] = load_tile16(Abase + (size_t)rowcol * KD + s * 32 + koff);

  unsigned long long rkey[8];
#pragma unroll
  for (int r = 0; r < 8; ++r) rkey[r] = 0ull;   // below any real packed key

  // B base already includes this lane's column offset (col = lbase + rowcol)
  const unsigned short* Bmat = PANn + ((size_t)b * L_ + rowcol) * KD;
  const int l0 = chunk * LT_PER_CHUNK * 16;

  v16bf B0[KSTEPS], B1[KSTEPS];
  loadB9(Bmat + (size_t)l0 * KD, koff, B0);

  for (int tt = 0; tt < LT_PER_CHUNK / 2; ++tt) {
    const int lb0 = l0 + (2 * tt) * 16;
    const int lb1 = lb0 + 16;
    const int lb2 = (2 * tt + 2 < LT_PER_CHUNK) ? lb1 + 16 : lb1;  // clamp (redundant last load)

    loadB9(Bmat + (size_t)lb1 * KD, koff, B1);   // overlap with B0's WMMAs
    tile_compute(Areg, B0, lb0, rowcol, rkey);

    loadB9(Bmat + (size_t)lb2 * KD, koff, B0);   // overlap with B1's WMMAs
    tile_compute(Areg, B1, lb1, rowcol, rkey);
  }

  // One cross-lane butterfly at the end: reduce 16 columns per half-wave.
  // C layout: reg r, lanes 0-15 -> (M=r, N=lane); lanes 16-31 -> (M=r+8, N=lane-16);
  // masks 1,2,4,8 never cross bit 4, so each half reduces independently.
#pragma unroll
  for (int r = 0; r < 8; ++r) {
    unsigned long long k = rkey[r];
#pragma unroll
    for (int msk = 1; msk < 16; msk <<= 1) {
      unsigned long long ok = __shfl_xor(k, msk, 32);
      k = (ok > k) ? ok : k;
    }
    rkey[r] = k;
  }

  if (rowcol == 0) {   // lane 0 -> rows 0..7, lane 16 -> rows 8..15
#pragma unroll
    for (int r = 0; r < 8; ++r) {
      int m = mbase + r + halfsel * 8;
      size_t o = ((size_t)b * NCHUNK + chunk) * L_ + m;
      pMax[o] = funmap((unsigned int)(rkey[r] >> 32));
      pArg[o] = (int)~((unsigned int)rkey[r]);
    }
  }
}

// ---------------------------------------------------------------------------
// Kernel 3: reduce the NCHUNK partial (max,arg) pairs (chunks in increasing l,
// strict > keeps first-occurrence argmax semantics across chunks).
// ---------------------------------------------------------------------------
__global__ __launch_bounds__(256)
void tf_reduce_chunks(const float* __restrict__ pMax, const int* __restrict__ pArg,
                      float* __restrict__ Rstar, int* __restrict__ Rarg) {
  int idx = blockIdx.x * blockDim.x + threadIdx.x;
  if (idx >= B_ * L_) return;
  int b = idx / L_, m = idx - b * L_;
  float best = -3.4e38f; int bi = 0;
  for (int c = 0; c < NCHUNK; ++c) {
    size_t o = ((size_t)b * NCHUNK + c) * L_ + m;
    float v = pMax[o];
    if (v > best) { best = v; bi = pArg[o]; }
  }
  Rstar[idx] = best;
  Rarg[idx]  = bi;
}

// ---------------------------------------------------------------------------
// Kernel 4: T_lv3 = fold3x3(gather columns of unfold(PAN) by Rarg) / 9,
// gathered directly from in_PAN (never materializing the unfold).
// ---------------------------------------------------------------------------
__global__ __launch_bounds__(256)
void tf_fold_gather(const float* __restrict__ PAN, const int* __restrict__ Rarg,
                    float* __restrict__ Tlv3) {
  int idx = blockIdx.x * blockDim.x + threadIdx.x;
  if (idx >= B_ * C_ * L_) return;
  int X = idx % W_;
  int Y = (idx / W_) % H_;
  int c = (idx / L_) % C_;
  int b = idx / (C_ * L_);
  const float* pb = PAN + (size_t)(b * C_ + c) * L_;
  const int*   ab = Rarg + (size_t)b * L_;
  float acc = 0.f;
#pragma unroll
  for (int i = 0; i < 3; ++i) {
    int y = Y + 1 - i;
    if (y < 0 || y >= H_) continue;
#pragma unroll
    for (int j = 0; j < 3; ++j) {
      int x = X + 1 - j;
      if (x < 0 || x >= W_) continue;
      int a  = ab[y * W_ + x];
      int ay = a / W_, ax = a - (a / W_) * W_;
      int py = ay + i - 1, px = ax + j - 1;
      if (py >= 0 && py < H_ && px >= 0 && px < W_)
        acc += pb[py * W_ + px];
    }
  }
  Tlv3[idx] = acc * (1.f / 9.f);
}

// ---------------------------------------------------------------------------
// Kernel 5: h = relu(conv3x3(concat([T_lv3, HSI]), w1, b1))
// ---------------------------------------------------------------------------
__global__ __launch_bounds__(256)
void tf_conv1_relu(const float* __restrict__ Tlv3, const float* __restrict__ HSI,
                   const float* __restrict__ w1, const float* __restrict__ b1,
                   float* __restrict__ hout) {
  int idx = blockIdx.x * blockDim.x + threadIdx.x;
  if (idx >= B_ * C_ * L_) return;
  int x  = idx % W_;
  int y  = (idx / W_) % H_;
  int co = (idx / L_) % C_;
  int b  = idx / (C_ * L_);
  float acc = b1[co];
  const float* wbase = w1 + (size_t)co * (2 * C_ * 9);
  for (int ci = 0; ci < 2 * C_; ++ci) {
    const float* src = (ci < C_) ? (Tlv3 + (size_t)(b * C_ + ci) * L_)
                                 : (HSI  + (size_t)(b * C_ + (ci - C_)) * L_);
    const float* wk = wbase + ci * 9;
#pragma unroll
    for (int ky = 0; ky < 3; ++ky) {
      int yy = y + ky - 1;
      if (yy < 0 || yy >= H_) continue;
#pragma unroll
      for (int kx = 0; kx < 3; ++kx) {
        int xx = x + kx - 1;
        if (xx < 0 || xx >= W_) continue;
        acc = fmaf(wk[ky * 3 + kx], src[yy * W_ + xx], acc);
      }
    }
  }
  hout[idx] = fmaxf(acc, 0.f);
}

// ---------------------------------------------------------------------------
// Kernel 6: out = conv3x3(h, w2, b2) * S + HSI
// ---------------------------------------------------------------------------
__global__ __launch_bounds__(256)
void tf_conv2_out(const float* __restrict__ hin, const float* __restrict__ w2,
                  const float* __restrict__ b2, const float* __restrict__ Rstar,
                  const float* __restrict__ HSI, float* __restrict__ out) {
  int idx = blockIdx.x * blockDim.x + threadIdx.x;
  if (idx >= B_ * C_ * L_) return;
  int x  = idx % W_;
  int y  = (idx / W_) % H_;
  int co = (idx / L_) % C_;
  int b  = idx / (C_ * L_);
  float acc = b2[co];
  const float* wbase = w2 + (size_t)co * (C_ * 9);
  for (int ci = 0; ci < C_; ++ci) {
    const float* src = hin + (size_t)(b * C_ + ci) * L_;
    const float* wk = wbase + ci * 9;
#pragma unroll
    for (int ky = 0; ky < 3; ++ky) {
      int yy = y + ky - 1;
      if (yy < 0 || yy >= H_) continue;
#pragma unroll
      for (int kx = 0; kx < 3; ++kx) {
        int xx = x + kx - 1;
        if (xx < 0 || xx >= W_) continue;
        acc = fmaf(wk[ky * 3 + kx], src[yy * W_ + xx], acc);
      }
    }
  }
  float s = Rstar[(size_t)b * L_ + y * W_ + x];
  out[idx] = acc * s + HSI[idx];
}

// ---------------------------------------------------------------------------
extern "C" void kernel_launch(void* const* d_in, const int* in_sizes, int n_in,
                              void* d_out, int out_size, void* d_ws, size_t ws_size,
                              hipStream_t stream) {
  (void)in_sizes; (void)n_in; (void)out_size; (void)ws_size;
  const float* in_HSI = (const float*)d_in[0];
  const float* in_PAN = (const float*)d_in[1];
  const float* w1     = (const float*)d_in[2];
  const float* b1     = (const float*)d_in[3];
  const float* w2     = (const float*)d_in[4];
  const float* b2     = (const float*)d_in[5];
  float* out = (float*)d_out;

  char* ws = (char*)d_ws;
  size_t off = 0;
  auto carve = [&](size_t bytes) -> char* {
    char* p = ws + off;
    off = (off + bytes + 255) & ~(size_t)255;
    return p;
  };
  unsigned short* HSIn = (unsigned short*)carve((size_t)B_ * L_ * KD * 2);      // 7.4 MB
  unsigned short* PANn = (unsigned short*)carve((size_t)B_ * L_ * KD * 2);      // 7.4 MB
  float* pMax  = (float*)carve((size_t)B_ * NCHUNK * L_ * 4);                   // 0.4 MB
  int*   pArg  = (int*)  carve((size_t)B_ * NCHUNK * L_ * 4);                   // 0.4 MB
  float* Rstar = (float*)carve((size_t)B_ * L_ * 4);
  int*   Rarg  = (int*)  carve((size_t)B_ * L_ * 4);
  float* Tlv3  = (float*)carve((size_t)B_ * C_ * L_ * 4);                       // 1.6 MB
  float* hbuf  = (float*)carve((size_t)B_ * C_ * L_ * 4);                       // 1.6 MB

  const int NPIX = B_ * C_ * L_;      // 409600
  tf_build_norm   <<<(B_ * L_ + 255) / 256, 256, 0, stream>>>(in_HSI, in_PAN, HSIn, PANn);
  tf_gemm_argmax  <<<(B_ * MTILES * NCHUNK) / 8, 256, 0, stream>>>(HSIn, PANn, pMax, pArg);
  tf_reduce_chunks<<<(B_ * L_ + 255) / 256, 256, 0, stream>>>(pMax, pArg, Rstar, Rarg);
  tf_fold_gather  <<<(NPIX + 255) / 256, 256, 0, stream>>>(in_PAN, Rarg, Tlv3);
  tf_conv1_relu   <<<(NPIX + 255) / 256, 256, 0, stream>>>(Tlv3, in_HSI, w1, b1, hbuf);
  tf_conv2_out    <<<(NPIX + 255) / 256, 256, 0, stream>>>(hbuf, w2, b2, Rstar, in_HSI, out);
}
